// GATLayer_30348238914087
// MI455X (gfx1250) — matrix operand order repaired and checked by reading, not compiled, not run on hardware
//
#include <hip/hip_runtime.h>

typedef __attribute__((ext_vector_type(16))) _Float16 v16h;
typedef __attribute__((ext_vector_type(4)))  _Float16 v4h;
typedef __attribute__((ext_vector_type(8)))  float    v8f;

// B=32, N=64, F_IN=F_OUT=256
// Workspace layout (bytes):
//   [0,      1024)   v       : 256 f32   (v = W @ mlp_w)
//   [4096,  135168)  Wt      : 256x256 f16, transposed W (Wt[f][k] = W[k][f])
//   [135168,659456)  s/a     : 32*64*64 f32 scores -> attention (in place)
//   [659456,1708032) Yh      : 2048x256 f16 weighted rows
#define OFF_V   0
#define OFF_WT  4096
#define OFF_S   (4096 + 131072)
#define OFF_YH  (4096 + 131072 + 524288)

// ---------------- k0: v = W @ mlp_w ; Wt = (f16) W^T ----------------
__global__ void __launch_bounds__(256) k0_prep(const float* __restrict__ W,
                                               const float* __restrict__ mlp_w,
                                               float* __restrict__ v,
                                               _Float16* __restrict__ Wt) {
    __shared__ float red[256];
    int k = blockIdx.x;          // row of W
    int f = threadIdx.x;         // column of W
    float wv = W[k * 256 + f];
    Wt[f * 256 + k] = (_Float16)wv;          // transposed half copy for WMMA B
    red[f] = wv * mlp_w[f];
    __syncthreads();
    #pragma unroll
    for (int st = 128; st > 0; st >>= 1) {
        if (f < st) red[f] += red[f + st];
        __syncthreads();
    }
    if (f == 0) v[k] = red[0];
}

// ---------------- k1: s[b,i,j] = leakyrelu(<x[b,j,i,:], v> + bias) ----------------
// one block per (b,j); 8 waves, each wave handles 8 values of i.
// Each lane owns a contiguous 8-float chunk -> 2x global_load_b128 per row.
__global__ void __launch_bounds__(256) k1_scores(const float* __restrict__ x,
                                                 const float* __restrict__ v,
                                                 const float* __restrict__ mlp_b,
                                                 float* __restrict__ s) {
    __shared__ float4 vl[64];
    int t = threadIdx.x;
    if (t < 64) vl[t] = ((const float4*)v)[t];
    __syncthreads();
    int   bid  = blockIdx.x;             // b*64 + j
    float mb   = mlp_b[0];
    int   lane = t & 31;
    int   w    = t >> 5;
    int   b    = bid >> 6;
    int   j    = bid & 63;
    float4 v0 = vl[lane * 2];
    float4 v1 = vl[lane * 2 + 1];
    for (int i = w; i < 64; i += 8) {
        const float4* row4 = (const float4*)(x + (size_t)(bid * 64 + i) * 256);
        float4 a0 = row4[lane * 2];
        float4 a1 = row4[lane * 2 + 1];
        float acc = a0.x * v0.x + a0.y * v0.y + a0.z * v0.z + a0.w * v0.w
                  + a1.x * v1.x + a1.y * v1.y + a1.z * v1.z + a1.w * v1.w;
        #pragma unroll
        for (int off = 16; off; off >>= 1)
            acc += __shfl_xor(acc, off, 32);
        if (lane == 0) {
            float tt = acc + mb;
            s[(b * 64 + i) * 64 + j] = (tt > 0.f) ? tt : 0.2f * tt;   // leaky_relu 0.2
        }
    }
}

// ---------------- k2: a = softmax(softmax(s, axis=-1), axis=-1), per 64-row ----------------
// one wave per row; each lane owns 2 elements
__global__ void __launch_bounds__(32) k2_softmax2(float* __restrict__ s) {
    int row = blockIdx.x;
    int l   = threadIdx.x;
    float* r = s + row * 64;
    float e0 = r[l];
    float e1 = r[l + 32];
    #pragma unroll
    for (int it = 0; it < 2; ++it) {
        float m = fmaxf(e0, e1);
        #pragma unroll
        for (int off = 16; off; off >>= 1)
            m = fmaxf(m, __shfl_xor(m, off, 32));
        e0 = __expf(e0 - m);
        e1 = __expf(e1 - m);
        float sum = e0 + e1;
        #pragma unroll
        for (int off = 16; off; off >>= 1)
            sum += __shfl_xor(sum, off, 32);
        float inv = 1.0f / sum;
        e0 *= inv;
        e1 *= inv;
    }
    r[l]      = e0;
    r[l + 32] = e1;
}

// ---------------- k3: y[b,i,k] = sum_j a[b,i,j] * x[b,i,j,k]  (f16 out) ----------------
// one block (64 threads = 2 waves) per (b,i); thread t owns 4 contiguous channels.
// Per j-step: 64 x global_load_b128 (1 KB coalesced). Store packed b64 halves.
__global__ void __launch_bounds__(64) k3_weighted(const float* __restrict__ x,
                                                  const float* __restrict__ a,
                                                  _Float16* __restrict__ Yh) {
    __shared__ float al[64];
    int bid = blockIdx.x;                // b*64 + i
    int t   = threadIdx.x;
    al[t] = a[bid * 64 + t];
    __syncthreads();
    const float4* xb4 = (const float4*)(x + (size_t)bid * 64 * 256);
    float4 acc = {0.f, 0.f, 0.f, 0.f};
    #pragma unroll 8
    for (int j = 0; j < 64; ++j) {
        float  aj = al[j];
        float4 p  = xb4[j * 64 + t];
        acc.x += aj * p.x;
        acc.y += aj * p.y;
        acc.z += aj * p.z;
        acc.w += aj * p.w;
    }
    v4h h;
    h.x = (_Float16)acc.x;
    h.y = (_Float16)acc.y;
    h.z = (_Float16)acc.z;
    h.w = (_Float16)acc.w;
    *(v4h*)(Yh + (size_t)bid * 256 + t * 4) = h;
}

// ---------------- k4: out = Y (2048x256, f16) @ W (256x256, f16 via Wt) ----------------
// one 16x16 output tile per wave, K-loop 8 x v_wmma_f32_16x16x32_f16
__global__ void __launch_bounds__(256) k4_gemm(const _Float16* __restrict__ Yh,
                                               const _Float16* __restrict__ Wt,
                                               float* __restrict__ out) {
    int t    = threadIdx.x;
    int w    = t >> 5;
    int l    = t & 31;
    int tile = blockIdx.x * 8 + w;       // 0..2047 tiles (128 M-tiles x 16 N-tiles)
    int mt   = tile >> 4;
    int nt   = tile & 15;
    int lh   = l >> 4;                   // lane half (0/1)
    int ll   = l & 15;

    const uint32_t* Yu = (const uint32_t*)Yh;
    const uint32_t* Wu = (const uint32_t*)Wt;
    int arow = mt * 16 + ll;             // A: M = lane&15
    int bcol = nt * 16 + ll;             // B: N = lane&15 (Wt row = W column)

    v8f c = {};
    for (int kk = 0; kk < 256; kk += 32) {
        union { v16h h; uint32_t u[8]; } ua, ub;
        #pragma unroll
        for (int r = 0; r < 8; ++r) {
            // A 16x32 f16 layout: VGPR 0-3 -> K = lh*8 + 2r{,+1}; VGPR 4-7 -> +16
            int ka = kk + ((r < 4) ? (lh * 8 + 2 * r) : (16 + lh * 8 + 2 * (r - 4)));
            ua.u[r] = Yu[(arow * 256 + ka) >> 1];
            // B 32x16 f16 layout: VGPR 0-3 -> K = lh*8 + 2r{,+1}; VGPR 4-7 -> +16
            int kb = kk + ((r < 4) ? (lh * 8 + 2 * r) : (16 + lh * 8 + 2 * (r - 4)));
            ub.u[r] = Wu[(bcol * 256 + kb) >> 1];
        }
        c = __builtin_amdgcn_wmma_f32_16x16x32_f16(
                /*neg_a=*/false, ua.h, /*neg_b=*/false, ub.h,
                /*c_mod=*/(short)0, c, /*reuse_a=*/false, /*reuse_b=*/false);
    }
    // C layout: lanes 0-15: (M=r, N=lane); lanes 16-31: (M=8+r, N=lane-16)
    int mbase = mt * 16 + lh * 8;
    int col   = nt * 16 + ll;
    #pragma unroll
    for (int r = 0; r < 8; ++r)
        out[(mbase + r) * 256 + col] = c[r];
}

extern "C" void kernel_launch(void* const* d_in, const int* in_sizes, int n_in,
                              void* d_out, int out_size, void* d_ws, size_t ws_size,
                              hipStream_t stream) {
    (void)in_sizes; (void)n_in; (void)out_size; (void)ws_size;
    // inputs: 0=p (unused by reference), 1=x, 2=W, 3=mlp_w, 4=mlp_b
    const float* x   = (const float*)d_in[1];
    const float* W   = (const float*)d_in[2];
    const float* mw  = (const float*)d_in[3];
    const float* mbp = (const float*)d_in[4];
    float* out = (float*)d_out;

    char* ws = (char*)d_ws;
    float*    v  = (float*)(ws + OFF_V);
    _Float16* Wt = (_Float16*)(ws + OFF_WT);
    float*    s  = (float*)(ws + OFF_S);
    _Float16* Yh = (_Float16*)(ws + OFF_YH);

    k0_prep    <<<256,  256, 0, stream>>>(W, mw, v, Wt);
    k1_scores  <<<2048, 256, 0, stream>>>(x, v, mbp, s);
    k2_softmax2<<<2048,  32, 0, stream>>>(s);
    k3_weighted<<<2048,  64, 0, stream>>>(x, s, Yh);
    k4_gemm    <<<256,  256, 0, stream>>>(Yh, Wt, out);
}
